// DropBlock_86861418594694
// MI455X (gfx1250) — compile-verified, matrix operand order checked by reading, not run on hardware
//
#include <hip/hip_runtime.h>
#include <hip/hip_bf16.h>
#include <stdint.h>

// Geometry (fixed by reference): B=32, C=256, H=W=64, block=5 -> mask 60x60
#define BS        5
#define H_        64
#define W_        64
#define MH        60          // H - BS + 1
#define MW        60          // W - BS + 1
#define PLANE_PIX 4096        // 64*64
#define PLANE_MSK 3600        // 60*60
#define NPLANES   8192        // 32*256
#define COUNT_M   33554432.0f // 8192*4096

typedef __attribute__((ext_vector_type(16))) _Float16 v16h;
typedef __attribute__((ext_vector_type(8)))  float    v8f;
typedef __attribute__((ext_vector_type(4)))  int      v4i;

#define AS1 __attribute__((address_space(1)))
#define AS3 __attribute__((address_space(3)))

#if defined(__HIP_DEVICE_COMPILE__) && __has_builtin(__builtin_amdgcn_global_load_async_to_lds_b128)
#define USE_ASYNC_LDS 1
#endif

// Load one 60x60 mask plane into LDS (padded stride 64) and compute the
// 5-wide row-max into s_rowmax (60x64). Uses CDNA5 async global->LDS DMA
// (ASYNCcnt) when available; falls back to float4 load + ds_store.
__device__ __forceinline__ void load_and_rowmax(const float* __restrict__ gmask,
                                                float* s_mask, float* s_rowmax) {
    const int tid = threadIdx.x;

    // 3600 floats = 900 x 16B chunks; each 60-float row = exactly 15 chunks,
    // so no chunk straddles the padded (stride-64) LDS rows.
    for (int i4 = tid; i4 < 900; i4 += 256) {
        const int row = i4 / 15;
        const int c4  = i4 - row * 15;
        float*       dst = s_mask + row * 64 + c4 * 4;
        const float* src = gmask + i4 * 4;
#ifdef USE_ASYNC_LDS
        __builtin_amdgcn_global_load_async_to_lds_b128((AS1 v4i*)src, (AS3 v4i*)dst,
                                                       /*offset=*/0, /*cpol=*/0);
#else
        *(float4*)dst = *(const float4*)src;
#endif
    }
#ifdef USE_ASYNC_LDS
#if __has_builtin(__builtin_amdgcn_s_wait_asynccnt)
    __builtin_amdgcn_s_wait_asynccnt(0);
#else
    asm volatile("s_wait_asynccnt 0" ::: "memory");
#endif
#endif
    __syncthreads();

    // Row-max: rowmax[h][w] = max_{j in [w-4, w] ∩ [0,60)} mask[h][j]
    for (int i = tid; i < MH * 64; i += 256) {
        const int h   = i >> 6;
        const int w   = i & 63;
        const int jlo = (w > BS - 1) ? (w - (BS - 1)) : 0;
        const int jhi = (w < MW) ? w : (MW - 1);
        float m = 0.0f;
        for (int j = jlo; j <= jhi; ++j) m = fmaxf(m, s_mask[h * 64 + j]);
        s_rowmax[i] = m;
    }
    __syncthreads();
}

__global__ void __launch_bounds__(256) zero_kernel(unsigned* cnt) { *cnt = 0u; }

// Pass 1: count ones of block_mask. Per-wave WMMA reduction:
// A = 16 block_mask values per lane (f16, exact 0/1), B = ones -> D rows hold
// lane-pair sums; sum 8 accumulators + shfl_xor(16) = exact wave total (<=512).
__global__ void __launch_bounds__(256) count_kernel(const float* __restrict__ mask,
                                                    unsigned* __restrict__ cnt) {
    __shared__ __align__(16) float s_mask[MH * 64];
    __shared__ __align__(16) float s_rowmax[MH * 64];

    const float* gmask = mask + (size_t)blockIdx.x * PLANE_MSK;
    load_and_rowmax(gmask, s_mask, s_rowmax);

    const int tid = threadIdx.x;
    v16h a;
#pragma unroll
    for (int k = 0; k < 4; ++k) {
        const int base = k * 1024 + tid * 4;   // 4 consecutive pixels, B128 aligned
        const int h    = base >> 6;
        const int w0   = base & 63;
        const int ilo  = (h > BS - 1) ? (h - (BS - 1)) : 0;
        const int ihi  = (h < MH) ? h : (MH - 1);
        float4 dv = make_float4(0.f, 0.f, 0.f, 0.f);
        for (int i = ilo; i <= ihi; ++i) {
            const float4 r = *(const float4*)(s_rowmax + i * 64 + w0);
            dv.x = fmaxf(dv.x, r.x); dv.y = fmaxf(dv.y, r.y);
            dv.z = fmaxf(dv.z, r.z); dv.w = fmaxf(dv.w, r.w);
        }
        a[4 * k + 0] = (_Float16)(1.0f - dv.x);
        a[4 * k + 1] = (_Float16)(1.0f - dv.y);
        a[4 * k + 2] = (_Float16)(1.0f - dv.z);
        a[4 * k + 3] = (_Float16)(1.0f - dv.w);
    }

    v16h bmat;
#pragma unroll
    for (int k = 0; k < 16; ++k) bmat[k] = (_Float16)1.0f;
    v8f c = {};
    // EXEC is all-ones here (no divergence since kernel entry in this path).
    c = __builtin_amdgcn_wmma_f32_16x16x32_f16(false, a, false, bmat,
                                               (short)0, c, false, false);
    float s = c[0] + c[1] + c[2] + c[3] + c[4] + c[5] + c[6] + c[7];
    s += __shfl_xor(s, 16, 32);            // lanes 0-15 hold M=0..7, 16-31 hold M=8..15
    if ((tid & 31) == 0)
        atomicAdd(cnt, (unsigned)(s + 0.5f));   // exact integer <= 512 per wave
}

// Pass 2: recompute dilation (mask re-read should hit the 192MB L2),
// apply out = (1 - dilated) * x * (countM / count). B128 in/out.
__global__ void __launch_bounds__(256) apply_kernel(const float* __restrict__ x,
                                                    const float* __restrict__ mask,
                                                    const unsigned* __restrict__ cnt,
                                                    float* __restrict__ out) {
    __shared__ __align__(16) float s_mask[MH * 64];
    __shared__ __align__(16) float s_rowmax[MH * 64];

    const size_t p     = blockIdx.x;
    const float* gmask = mask + p * PLANE_MSK;
    const float* gx    = x + p * PLANE_PIX;
    float*       gout  = out + p * PLANE_PIX;
    const float  scale = COUNT_M / (float)(*cnt);

    load_and_rowmax(gmask, s_mask, s_rowmax);

    const int tid = threadIdx.x;
#pragma unroll
    for (int k = 0; k < 4; ++k) {
        const int base = k * 1024 + tid * 4;
        const int h    = base >> 6;
        const int w0   = base & 63;
        const int ilo  = (h > BS - 1) ? (h - (BS - 1)) : 0;
        const int ihi  = (h < MH) ? h : (MH - 1);
        float4 dv = make_float4(0.f, 0.f, 0.f, 0.f);
        for (int i = ilo; i <= ihi; ++i) {
            const float4 r = *(const float4*)(s_rowmax + i * 64 + w0);
            dv.x = fmaxf(dv.x, r.x); dv.y = fmaxf(dv.y, r.y);
            dv.z = fmaxf(dv.z, r.z); dv.w = fmaxf(dv.w, r.w);
        }
        const float4 xv = *(const float4*)(gx + base);
        float4 o;
        o.x = (1.0f - dv.x) * xv.x * scale;
        o.y = (1.0f - dv.y) * xv.y * scale;
        o.z = (1.0f - dv.z) * xv.z * scale;
        o.w = (1.0f - dv.w) * xv.w * scale;
        *(float4*)(gout + base) = o;
    }
}

extern "C" void kernel_launch(void* const* d_in, const int* in_sizes, int n_in,
                              void* d_out, int out_size, void* d_ws, size_t ws_size,
                              hipStream_t stream) {
    const float* x    = (const float*)d_in[0];
    const float* mask = (const float*)d_in[1];
    float*       out  = (float*)d_out;
    unsigned*    cnt  = (unsigned*)d_ws;

    zero_kernel<<<1, 1, 0, stream>>>(cnt);
    count_kernel<<<NPLANES, 256, 0, stream>>>(mask, cnt);
    apply_kernel<<<NPLANES, 256, 0, stream>>>(x, mask, cnt, out);
}
